// LinearAttention_18837726560988
// MI455X (gfx1250) — matrix operand order repaired and checked by reading, not compiled, not run on hardware
//
#include <hip/hip_runtime.h>
#include <hip/hip_bf16.h>
#include <math.h>
#include <stdint.h>

// ---------------------------------------------------------------------------
// CDNA5 (gfx1250, wave32) types
// ---------------------------------------------------------------------------
typedef __attribute__((ext_vector_type(16))) __bf16 v16bf;
typedef __attribute__((ext_vector_type(8)))  __bf16 v8bf;
typedef __attribute__((ext_vector_type(4)))  __bf16 v4bf;
typedef __attribute__((ext_vector_type(8)))  float  v8f;
typedef __attribute__((ext_vector_type(4)))  int    v4i;

// LDS tile row stride in bf16 elements: 112B rows -> 16B aligned for b128 LDS
// ops AND (r*28)%64 hits 16 distinct bank groups for the 16 fragment rows.
#define LDSB 56

// ---------------------------------------------------------------------------
// CDNA5 async global->LDS copy (16B per lane), ASYNCcnt-tracked.
// Toolchain declares the builtin with (v4i AS1*, v4i AS3*, imm, imm) params.
// ---------------------------------------------------------------------------
__device__ __forceinline__ void async_copy_b128(const void* g, void* l)
{
#if __has_builtin(__builtin_amdgcn_global_load_async_to_lds_b128)
    __builtin_amdgcn_global_load_async_to_lds_b128(
        (__attribute__((address_space(1))) v4i*)(uintptr_t)g,
        (__attribute__((address_space(3))) v4i*)(uintptr_t)(uint32_t)(uintptr_t)l,
        0, 0);
#else
    asm volatile("global_load_async_to_lds_b128 %0, %1, off"
                 :: "v"((uint32_t)(uintptr_t)l), "v"((uint64_t)(uintptr_t)g)
                 : "memory");
#endif
}

template <int N>
__device__ __forceinline__ void wait_asynccnt()
{
#if __has_builtin(__builtin_amdgcn_s_wait_asynccnt)
    __builtin_amdgcn_s_wait_asynccnt(N);
#else
    asm volatile("s_wait_asynccnt %0" :: "i"(N) : "memory");
#endif
}

// ---------------------------------------------------------------------------
// Stage a 64-row x 32-col bf16 tile (4KB) row-major into LDS via async copies.
// 256 chunks of 16B; 128 threads -> 2 async issues per thread (2/wave ASYNCcnt).
// ---------------------------------------------------------------------------
__device__ __forceinline__ void issue_async_rowtile(
    const __bf16* __restrict__ g, int ld, int k0,
    __bf16 (*dst)[LDSB], int tid)
{
    #pragma unroll
    for (int i = 0; i < 2; ++i) {
        const int idx = i * 128 + tid;
        const int row = idx >> 2;          // 4 chunks per 32-elem row
        const int col = (idx & 3) * 8;
        async_copy_b128(g + (long)row * ld + k0 + col, &dst[row][col]);
    }
}

// B (K x N row-major) tile: pipelined through registers, stored transposed
// (Bs[n][k]) so fragment reads are contiguous b128s.
__device__ __forceinline__ void load_B_regs(
    const __bf16* __restrict__ B, int ldb, int k0, v8bf br[2], int tid)
{
    #pragma unroll
    for (int i = 0; i < 2; ++i) {
        const int idx  = i * 128 + tid;
        const int krow = idx >> 3;         // 8 chunks per 64-elem row
        const int nc   = (idx & 7) * 8;
        br[i] = *(const v8bf*)(B + (long)(k0 + krow) * ldb + nc);
    }
}

__device__ __forceinline__ void store_B_lds(
    const v8bf br[2], __bf16 (*Bs)[LDSB], int tid)
{
    #pragma unroll
    for (int i = 0; i < 2; ++i) {
        const int idx  = i * 128 + tid;
        const int krow = idx >> 3;
        const int nc   = (idx & 7) * 8;
        #pragma unroll
        for (int j = 0; j < 8; ++j) Bs[nc + j][krow] = br[i][j];
    }
}

// ---------------------------------------------------------------------------
// Fragment assembly + 4 back-to-back WMMAs. Layouts per ISA 7.12.2 (wave32):
//   A 16x32 bf16 : lane L -> M=L%16 ; VGPR v<4: K=(L>=16?8:0)+2v ; v>=4: +16
//   B 32x16 bf16 : lane L -> N=L%16 ; VGPR v<4: K=(L>=16?8:0)+2v ; v>=4: +16
//   C/D 16x16 f32: lane L -> N=L%16 ; VGPR r : M=r+(L>=16?8:0)
// ---------------------------------------------------------------------------
__device__ __forceinline__ void wmma_compute(
    const __bf16 (*As)[LDSB], const __bf16 (*Bs)[LDSB],
    v8f acc[4], int wave, int l16, int kh)
{
    v16bf a;
    const __bf16* ar = As[wave * 16 + l16];
    #pragma unroll
    for (int i = 0; i < 4; ++i) {
        const int k = kh * 8 + 2 * i;
        a[2 * i] = ar[k]; a[2 * i + 1] = ar[k + 1];
    }
    #pragma unroll
    for (int i = 0; i < 4; ++i) {
        const int k = 16 + kh * 8 + 2 * i;
        a[8 + 2 * i] = ar[k]; a[9 + 2 * i] = ar[k + 1];
    }
    v16bf bf[4];
    #pragma unroll
    for (int nt = 0; nt < 4; ++nt) {
        const __bf16* br = Bs[nt * 16 + l16];
        const int kb = kh * 8;
        #pragma unroll
        for (int i = 0; i < 8; ++i) {
            const int k = kb + 2 * i + (i >= 4 ? 8 : 0);
            bf[nt][2 * i] = br[k]; bf[nt][2 * i + 1] = br[k + 1];
        }
    }
    #pragma unroll
    for (int nt = 0; nt < 4; ++nt)
        acc[nt] = __builtin_amdgcn_wmma_f32_16x16x32_bf16(
            false, a, false, bf[nt], (short)0, acc[nt], false, false);
}

// ---------------------------------------------------------------------------
// Double-buffered 64x64-tile GEMM core, all-bf16 operands, fp32 accum.
//   BT=true : B is 64 x K row-major ("N-major") -> fully async staging
//   BT=false: B is K x N row-major (pre-offset to n0) -> register pipeline
// ---------------------------------------------------------------------------
template <bool BT>
__device__ __forceinline__ void gemm_tile_bf16(
    const __bf16* __restrict__ A, int lda,
    const __bf16* __restrict__ B, int ldb,
    int K,
    __bf16 (*As)[64][LDSB], __bf16 (*Bs)[64][LDSB],
    v8f acc[4], int tid)
{
    const int lane = tid & 31, wave = tid >> 5;
    const int l16 = lane & 15, kh = lane >> 4;

    v8bf breg[2], breg2[2];

    // prologue: tile 0
    issue_async_rowtile(A, lda, 0, As[0], tid);
    if (BT) issue_async_rowtile(B, ldb, 0, Bs[0], tid);
    else    load_B_regs(B, ldb, 0, breg, tid);

    const int nT = K >> 5;
    for (int t = 0; t < nT; ++t) {
        const int cur = t & 1, nxt = cur ^ 1;
        const bool more = (t + 1) < nT;
        if (more) {
            issue_async_rowtile(A, lda, (t + 1) * 32, As[nxt], tid);
            if (BT) issue_async_rowtile(B, ldb, (t + 1) * 32, Bs[nxt], tid);
            else    load_B_regs(B, ldb, (t + 1) * 32, breg2, tid);
        }
        // wait for *current* tile's async copies (in-order completion)
        if (BT) { if (more) wait_asynccnt<4>(); else wait_asynccnt<0>(); }
        else    { if (more) wait_asynccnt<2>(); else wait_asynccnt<0>(); }
        if (!BT) store_B_lds(breg, Bs[cur], tid);
        __syncthreads();
        wmma_compute(As[cur], Bs[cur], acc, wave, l16, kh);
        __syncthreads();
        if (!BT) { breg[0] = breg2[0]; breg[1] = breg2[1]; }
    }
}

// ---------------------------------------------------------------------------
// Projection GEMM: C[b] = W(MxK) @ B[b](KxN)  (+bias). bf16 in; out bf16 or f32.
//   grid = (N/64, M/64, batches), block = 128
// ---------------------------------------------------------------------------
template <bool OUT_BF16>
__global__ void __launch_bounds__(128) gemm_proj_kernel(
    const __bf16* __restrict__ W, const __bf16* __restrict__ Bm,
    void* __restrict__ Cm, const float* __restrict__ bias,
    int N, int K, long strideB, long strideC)
{
    __shared__ __bf16 As[2][64][LDSB];
    __shared__ __bf16 Bs[2][64][LDSB];

    const int n0 = blockIdx.x * 64, m0 = blockIdx.y * 64, b = blockIdx.z;
    const __bf16* A  = W + (long)m0 * K;
    const __bf16* Bp = Bm + (long)b * strideB + n0;

    v8f acc[4];
    #pragma unroll
    for (int i = 0; i < 4; ++i) acc[i] = (v8f){0.f,0.f,0.f,0.f,0.f,0.f,0.f,0.f};

    gemm_tile_bf16<false>(A, K, Bp, N, K, As, Bs, acc, threadIdx.x);

    const int lane = threadIdx.x & 31, wave = threadIdx.x >> 5;
    const int l16 = lane & 15, kh = lane >> 4;
    #pragma unroll
    for (int nt = 0; nt < 4; ++nt) {
        #pragma unroll
        for (int r = 0; r < 8; ++r) {
            const int row = m0 + wave * 16 + r + kh * 8;
            const int col = n0 + nt * 16 + l16;
            if (OUT_BF16) {
                __bf16* C = (__bf16*)Cm + (long)b * strideC;
                C[(long)row * N + col] = (__bf16)acc[nt][r];
            } else {
                float* C = (float*)Cm + (long)b * strideC;
                C[(long)row * N + col] = acc[nt][r] + bias[row];
            }
        }
    }
}

// ---------------------------------------------------------------------------
// Softmax over n=4096 on the K slice of bf16 qkv, in place. grid=8192, blk=256
// ---------------------------------------------------------------------------
__global__ void softmax_kernel(__bf16* __restrict__ qkv)
{
    const int r = blockIdx.x;
    const int b = r >> 9, j = r & 511;
    __bf16* row = qkv + ((long)(b * 1536 + 512 + j)) * 4096;

    __shared__ float red[256];
    const int t = threadIdx.x;

    float m = -INFINITY;
    for (int i = t; i < 4096; i += 256) m = fmaxf(m, (float)row[i]);
    red[t] = m; __syncthreads();
    for (int s = 128; s > 0; s >>= 1) {
        if (t < s) red[t] = fmaxf(red[t], red[t + s]);
        __syncthreads();
    }
    m = red[0]; __syncthreads();

    float sum = 0.f;
    for (int i = t; i < 4096; i += 256) sum += __expf((float)row[i] - m);
    red[t] = sum; __syncthreads();
    for (int s = 128; s > 0; s >>= 1) {
        if (t < s) red[t] += red[t + s];
        __syncthreads();
    }
    const float inv = 1.0f / red[0];
    for (int i = t; i < 4096; i += 256)
        row[i] = (__bf16)(__expf((float)row[i] - m) * inv);
}

// ---------------------------------------------------------------------------
// Context split-K: ctxT[bh][e][d] += sum_n Ksm[d][n] * V[e][n]
//   Both operands n-major -> fully async staging. grid=(8,128), blk=128.
// ---------------------------------------------------------------------------
__global__ void __launch_bounds__(128) context_kernel(
    const __bf16* __restrict__ qkv, float* __restrict__ ctxT)
{
    __shared__ __bf16 As[2][64][LDSB];
    __shared__ __bf16 Bs[2][64][LDSB];

    const int bh = blockIdx.y, b = bh >> 3, h = bh & 7;
    const int k0 = blockIdx.x * 512;

    const __bf16* A = qkv + ((long)(b * 1536 + 512  + h * 64)) * 4096 + k0; // Ksm
    const __bf16* B = qkv + ((long)(b * 1536 + 1024 + h * 64)) * 4096 + k0; // V

    v8f acc[4];
    #pragma unroll
    for (int i = 0; i < 4; ++i) acc[i] = (v8f){0.f,0.f,0.f,0.f,0.f,0.f,0.f,0.f};

    gemm_tile_bf16<true>(A, 4096, B, 4096, 512, As, Bs, acc, threadIdx.x);

    const int lane = threadIdx.x & 31, wave = threadIdx.x >> 5;
    const int l16 = lane & 15, kh = lane >> 4;
    float* base = ctxT + (long)bh * 4096;
    #pragma unroll
    for (int nt = 0; nt < 4; ++nt) {
        #pragma unroll
        for (int r = 0; r < 8; ++r) {
            const int d = wave * 16 + r + kh * 8;
            const int e = nt * 16 + l16;
            atomicAdd(&base[e * 64 + d], acc[nt][r]);   // store transposed
        }
    }
}

// ---------------------------------------------------------------------------
// Apply: attn[e][n] = sum_d ctxT[e][d] * Q[d][n]; bf16 out over dead K slice.
//   grid = (64, 8, 16), block = 128
// ---------------------------------------------------------------------------
__global__ void __launch_bounds__(128) apply_context_kernel(
    __bf16* __restrict__ qkv, const __bf16* __restrict__ ctxb)
{
    __shared__ __bf16 As[2][64][LDSB];
    __shared__ __bf16 Bs[2][64][LDSB];

    const int n0 = blockIdx.x * 64, h = blockIdx.y, b = blockIdx.z;
    const __bf16* A  = ctxb + ((long)(b * 8 + h)) * 4096;             // 64x64
    const __bf16* Bq = qkv + ((long)(b * 1536 + h * 64)) * 4096 + n0; // Q

    v8f acc[4];
    #pragma unroll
    for (int i = 0; i < 4; ++i) acc[i] = (v8f){0.f,0.f,0.f,0.f,0.f,0.f,0.f,0.f};

    gemm_tile_bf16<false>(A, 64, Bq, 4096, 64, As, Bs, acc, threadIdx.x);

    const int lane = threadIdx.x & 31, wave = threadIdx.x >> 5;
    const int l16 = lane & 15, kh = lane >> 4;
    __bf16* C = qkv + ((long)(b * 1536 + 512 + h * 64)) * 4096;
    #pragma unroll
    for (int nt = 0; nt < 4; ++nt) {
        #pragma unroll
        for (int r = 0; r < 8; ++r) {
            const int e   = wave * 16 + r + kh * 8;
            const int col = n0 + nt * 16 + l16;
            C[(long)e * 4096 + col] = (__bf16)acc[nt][r];
        }
    }
}

// ---------------------------------------------------------------------------
// Utility kernels
// ---------------------------------------------------------------------------
__global__ void conv_f32_bf16_kernel(const float* __restrict__ in,
                                     __bf16* __restrict__ out, int n4)
{
    const int i = blockIdx.x * blockDim.x + threadIdx.x;
    if (i < n4) {
        const float4 v = ((const float4*)in)[i];
        v4bf o; o[0] = (__bf16)v.x; o[1] = (__bf16)v.y;
        o[2] = (__bf16)v.z; o[3] = (__bf16)v.w;
        ((v4bf*)out)[i] = o;
    }
}

__global__ void zero_kernel(float* __restrict__ p, int n)
{
    const int i = blockIdx.x * blockDim.x + threadIdx.x;
    if (i < n) p[i] = 0.f;
}

// ---------------------------------------------------------------------------
// ws layout (bf16 unless noted):
//   qkvb  16*1536*4096            (192MB)
//   xbf   16*512*4096             ( 64MB)
//   wqkvb 1536*512, woutb 512*512 (  2MB)
//   ctxT  f32 16*8*64*64 (2MB) | ctxb bf16 (1MB)     total ~261MB
// ---------------------------------------------------------------------------
extern "C" void kernel_launch(void* const* d_in, const int* in_sizes, int n_in,
                              void* d_out, int out_size, void* d_ws, size_t ws_size,
                              hipStream_t stream)
{
    const float* x     = (const float*)d_in[0];
    const float* w_qkv = (const float*)d_in[1];
    const float* w_out = (const float*)d_in[2];
    const float* b_out = (const float*)d_in[3];
    float* out = (float*)d_out;

    __bf16* qkvb  = (__bf16*)d_ws;
    __bf16* xbf   = qkvb  + (size_t)16 * 1536 * 4096;
    __bf16* wqkvb = xbf   + (size_t)16 * 512 * 4096;
    __bf16* woutb = wqkvb + (size_t)1536 * 512;
    float*  ctxT  = (float*)(woutb + (size_t)512 * 512);
    __bf16* ctxb  = (__bf16*)(ctxT + 524288);

    // 0) one-time fp32 -> bf16 conversions + split-K accumulator clear
    conv_f32_bf16_kernel<<<(16*512*4096/4 + 255)/256, 256, 0, stream>>>(x, xbf, 16*512*4096/4);
    conv_f32_bf16_kernel<<<(1536*512/4    + 255)/256, 256, 0, stream>>>(w_qkv, wqkvb, 1536*512/4);
    conv_f32_bf16_kernel<<<(512*512/4     + 255)/256, 256, 0, stream>>>(w_out, woutb, 512*512/4);
    zero_kernel<<<(524288 + 255)/256, 256, 0, stream>>>(ctxT, 524288);

    // 1) qkv = W_qkv @ X  (bf16 out)
    gemm_proj_kernel<true><<<dim3(64, 24, 16), 128, 0, stream>>>(
        wqkvb, xbf, qkvb, nullptr, 4096, 512,
        (long)512 * 4096, (long)1536 * 4096);

    // 2) softmax over n on K slice (in place, bf16)
    softmax_kernel<<<16 * 512, 256, 0, stream>>>(qkvb);

    // 3) ctxT[e][d] = sum_n Ksm[d][n] V[e][n]  (split-K=8, f32 atomics)
    context_kernel<<<dim3(8, 128), 128, 0, stream>>>(qkvb, ctxT);

    // 3b) ctx -> bf16
    conv_f32_bf16_kernel<<<(524288/4 + 255)/256, 256, 0, stream>>>(ctxT, ctxb, 524288/4);

    // 4) attn[e][n] = sum_d ctx[e][d] Q[d][n]  (bf16 out over dead K slice)
    apply_context_kernel<<<dim3(64, 8, 16), 128, 0, stream>>>(qkvb, ctxb);

    // 5) out = W_out @ attn + b_out  (f32 out)
    gemm_proj_kernel<false><<<dim3(64, 8, 16), 128, 0, stream>>>(
        woutb, qkvb + (size_t)512 * 4096, out, b_out, 4096, 512,
        (long)1536 * 4096, (long)512 * 4096);
}
